// CASVDDenseMul_73426760892513
// MI455X (gfx1250) — matrix operand order repaired and checked by reading, not compiled, not run on hardware
//
#include <hip/hip_runtime.h>
#include <hip/hip_bf16.h>
#include <stdint.h>

// B_SZ=4096, N_IN=4096, N_CTX=512, UNITS=4096, RANK=256
#define B_SZ  4096
#define N_IN  4096
#define N_CTX 512
#define UNITS 4096
#define RANK  256

typedef __attribute__((ext_vector_type(16))) __bf16 v16bf;
typedef __attribute__((ext_vector_type(8)))  float  v8f;
typedef __attribute__((ext_vector_type(4)))  int    v4i;

typedef __attribute__((address_space(1))) v4i* gv4i_ptr;  // global
typedef __attribute__((address_space(3))) v4i* lv4i_ptr;  // LDS

union FragB16 { v16bf v; uint32_t d[8]; };

__device__ __forceinline__ v8f wmma_bf16(v16bf a, v16bf b, v8f c) {
  // (neg_a, A, neg_b, B, c_mod, C, reuse_a, reuse_b)
  return __builtin_amdgcn_wmma_f32_16x16x32_bf16(false, a, false, b, (short)0, c,
                                                 false, false);
}

// ---- CDNA5 async global->LDS (ASYNCcnt-tracked, bypasses VGPRs) ----
__device__ __forceinline__ void async_copy_b128(const void* g, void* l) {
#if __has_builtin(__builtin_amdgcn_global_load_async_to_lds_b128)
  __builtin_amdgcn_global_load_async_to_lds_b128((gv4i_ptr)g, (lv4i_ptr)l, 0, 0);
#else
  *(uint4*)l = *(const uint4*)g;  // fallback: synchronous VGPR copy
#endif
}

__device__ __forceinline__ void wait_async() {
#if __has_builtin(__builtin_amdgcn_s_wait_asynccnt)
  __builtin_amdgcn_s_wait_asynccnt(0);
#else
  asm volatile("s_wait_asynccnt 0x0" ::: "memory");
#endif
}

// One K=32 step for one wave: A tile lA[64][32] (bf16, k-contig),
// B tile lB[256][32] (bf16, [n][k] so lane fragments are contiguous b128s).
__device__ __forceinline__ void wave_mma_step(const __bf16* lA, const __bf16* lB,
                                              int mbase, int nhalf, int lane,
                                              v8f acc[8]) {
  const int half = (lane >= 16) ? 1 : 0;
  const int m = mbase + (lane & 15);
  FragB16 a;
  const uint32_t* pa = (const uint32_t*)(lA + m * 32);
  // 16-bit A layout: lanes 0-15 hold K {0..7,16..23}; lanes 16-31 hold {8..15,24..31}
#pragma unroll
  for (int j = 0; j < 8; ++j) {
    int idx = (j < 4 ? j : j + 4) + 4 * half;
    a.d[j] = pa[idx];
  }
#pragma unroll
  for (int t = 0; t < 8; ++t) {
    const int n = nhalf + t * 16 + (lane & 15);
    const uint32_t* pb = (const uint32_t*)(lB + n * 32);
    FragB16 b;
    // B layout: lanes 0-15 hold K {0..7,16..23}; lanes 16-31 hold {8..15,24..31}
#pragma unroll
    for (int j = 0; j < 8; ++j) b.d[j] = pb[(j < 4 ? j : j + 4) + 4 * half];
    acc[t] = wmma_bf16(a.v, b.v, acc[t]);
  }
}

// Stage a [256 n][32 k] bf16 B tile from a pre-packed [n][Kt] bf16 array (async).
__device__ __forceinline__ void stage_B_async(const __bf16* Bt, int Kt, int k0,
                                              __bf16* lB, int t) {
#pragma unroll
  for (int i = 0; i < 4; ++i) {
    int idx = t + i * 256;
    int n = idx >> 2, j = idx & 3;  // 4 x 16B chunks per 64B row
    async_copy_b128(Bt + (size_t)n * Kt + k0 + j * 8, lB + n * 32 + j * 8);
  }
}

// Stage a [64 m][32 k] tile of streamed f32 A with in-register f32->bf16 convert.
__device__ __forceinline__ void stage_A_cvt(const float* A, int Kt, int rowBlock,
                                            int k0, __bf16* lA, int t) {
#pragma unroll
  for (int i = 0; i < 8; ++i) {
    int idx = t + i * 256;
    int r = idx >> 5, k = idx & 31;
    lA[r * 32 + k] = (__bf16)A[(size_t)(rowBlock + r) * Kt + k0 + k];
  }
}

// ------------- one-shot weight packing: U->Ut[n][k], W->Wt[n][k], V->Vb -------------
__global__ __launch_bounds__(256) void pack_weights(const float* __restrict__ U,
                                                    const float* __restrict__ W,
                                                    const float* __restrict__ V,
                                                    __bf16* __restrict__ Ut,
                                                    __bf16* __restrict__ Wt,
                                                    __bf16* __restrict__ Vb) {
  int idx = blockIdx.x * 256 + threadIdx.x;
  if (idx < N_IN * RANK) {           // U[k][n] -> Ut[n][k]
    int k = idx >> 8, n = idx & 255;
    Ut[(size_t)n * N_IN + k] = (__bf16)U[idx];
  }
  if (idx < N_CTX * RANK) {          // W[k][n] -> Wt[n][k]
    int k = idx >> 8, n = idx & 255;
    Wt[(size_t)n * N_CTX + k] = (__bf16)W[idx];
  }
  if (idx < UNITS * RANK) {          // V[m][r] is already [n][k] for t@V^T
    Vb[idx] = (__bf16)V[idx];
  }
}

// -------------------- kernel 1: s = S * sigmoid(context @ W + B) --------------------
__global__ __launch_bounds__(256) void s_kernel(const float* __restrict__ ctx,
                                                const __bf16* __restrict__ Wt,
                                                const float* __restrict__ Bv,
                                                const float* __restrict__ S,
                                                float* __restrict__ s_out) {
  __shared__ __attribute__((aligned(16))) __bf16 lA[2][64 * 32];
  __shared__ __attribute__((aligned(16))) __bf16 lB[2][RANK * 32];
  const int t = threadIdx.x;
  const int lane = t & 31, wave = t >> 5;
  const int mbase = (wave & 3) * 16, nhalf = (wave >> 2) * 128;
  const int rowBlock = blockIdx.x * 64;
  const int NSTEP = N_CTX / 32;

  v8f acc[8];
  const v8f zero = {0.f, 0.f, 0.f, 0.f, 0.f, 0.f, 0.f, 0.f};
#pragma unroll
  for (int i = 0; i < 8; ++i) acc[i] = zero;

  stage_B_async(Wt, N_CTX, 0, lB[0], t);
  stage_A_cvt(ctx, N_CTX, rowBlock, 0, lA[0], t);
  wait_async();
  __syncthreads();

  for (int i = 0; i < NSTEP; ++i) {
    const int b = i & 1;
    if (i + 1 < NSTEP) {  // prefetch next step into the other buffer
      stage_B_async(Wt, N_CTX, (i + 1) * 32, lB[b ^ 1], t);
      stage_A_cvt(ctx, N_CTX, rowBlock, (i + 1) * 32, lA[b ^ 1], t);
    }
    wave_mma_step(lA[b], lB[b], mbase, nhalf, lane, acc);
    wait_async();
    __syncthreads();
  }

  const int half = (lane >= 16) ? 1 : 0;
#pragma unroll
  for (int tt = 0; tt < 8; ++tt) {
    const int n = nhalf + tt * 16 + (lane & 15);
    const float sv = S[n], bv = Bv[n];
#pragma unroll
    for (int i = 0; i < 8; ++i) {
      const int m = rowBlock + mbase + i + 8 * half;
      const float x = acc[tt][i] + bv;
      s_out[(size_t)m * RANK + n] = sv / (1.0f + __expf(-x));
    }
  }
}

// -------------------- kernel 2: t = (inputs @ U) * s, stored bf16 --------------------
__global__ __launch_bounds__(256) void t_kernel(const float* __restrict__ A,
                                                const __bf16* __restrict__ Ut,
                                                const float* __restrict__ s_in,
                                                __bf16* __restrict__ t_out) {
  __shared__ __attribute__((aligned(16))) __bf16 lA[2][64 * 32];
  __shared__ __attribute__((aligned(16))) __bf16 lB[2][RANK * 32];
  const int t = threadIdx.x;
  const int lane = t & 31, wave = t >> 5;
  const int mbase = (wave & 3) * 16, nhalf = (wave >> 2) * 128;
  const int rowBlock = blockIdx.x * 64;
  const int NSTEP = N_IN / 32;

  v8f acc[8];
  const v8f zero = {0.f, 0.f, 0.f, 0.f, 0.f, 0.f, 0.f, 0.f};
#pragma unroll
  for (int i = 0; i < 8; ++i) acc[i] = zero;

  stage_B_async(Ut, N_IN, 0, lB[0], t);
  stage_A_cvt(A, N_IN, rowBlock, 0, lA[0], t);
  wait_async();
  __syncthreads();

  for (int i = 0; i < NSTEP; ++i) {
    const int b = i & 1;
    if (i + 1 < NSTEP) {
      stage_B_async(Ut, N_IN, (i + 1) * 32, lB[b ^ 1], t);
      stage_A_cvt(A, N_IN, rowBlock, (i + 1) * 32, lA[b ^ 1], t);
      // prefetch the streamed input two steps ahead (global_prefetch_b8)
      if (i + 2 < NSTEP)
        __builtin_prefetch(&A[(size_t)(rowBlock + (t >> 5) * 8) * N_IN + (i + 2) * 32], 0, 1);
    }
    wave_mma_step(lA[b], lB[b], mbase, nhalf, lane, acc);
    wait_async();
    __syncthreads();
  }

  const int half = (lane >= 16) ? 1 : 0;
#pragma unroll
  for (int tt = 0; tt < 8; ++tt) {
    const int n = nhalf + tt * 16 + (lane & 15);
#pragma unroll
    for (int i = 0; i < 8; ++i) {
      const int m = rowBlock + mbase + i + 8 * half;
      const float v = acc[tt][i] * s_in[(size_t)m * RANK + n];
      t_out[(size_t)m * RANK + n] = (__bf16)v;
    }
  }
}

// -------------------- kernel 3: out = relu(t @ V^T + 2*bias) --------------------
__global__ __launch_bounds__(256) void out_kernel(const __bf16* __restrict__ t_in,
                                                  const __bf16* __restrict__ Vb,
                                                  const float* __restrict__ bias,
                                                  float* __restrict__ out) {
  __shared__ __attribute__((aligned(16))) __bf16 lA[2][64 * 32];
  __shared__ __attribute__((aligned(16))) __bf16 lB[2][256 * 32];
  const int t = threadIdx.x;
  const int lane = t & 31, wave = t >> 5;
  const int mbase = (wave & 3) * 16, nhalf = (wave >> 2) * 128;
  const int rowBlock = blockIdx.y * 64;
  const int colBlock = blockIdx.x * 256;
  const int NSTEP = RANK / 32;

  v8f acc[8];
  const v8f zero = {0.f, 0.f, 0.f, 0.f, 0.f, 0.f, 0.f, 0.f};
#pragma unroll
  for (int i = 0; i < 8; ++i) acc[i] = zero;

  const __bf16* Vcol = Vb + (size_t)colBlock * RANK;

  // A tile (t, already bf16): 64 rows x 64B = 256 b128 => exactly 1 per thread
  {
    int r = t >> 2, j = t & 3;
    async_copy_b128(t_in + (size_t)(rowBlock + r) * RANK + j * 8, lA[0] + r * 32 + j * 8);
  }
  stage_B_async(Vcol, RANK, 0, lB[0], t);
  wait_async();
  __syncthreads();

  for (int i = 0; i < NSTEP; ++i) {
    const int b = i & 1;
    if (i + 1 < NSTEP) {
      int k0 = (i + 1) * 32;
      int r = t >> 2, j = t & 3;
      async_copy_b128(t_in + (size_t)(rowBlock + r) * RANK + k0 + j * 8,
                      lA[b ^ 1] + r * 32 + j * 8);
      stage_B_async(Vcol, RANK, k0, lB[b ^ 1], t);
    }
    wave_mma_step(lA[b], lB[b], mbase, nhalf, lane, acc);
    wait_async();
    __syncthreads();
  }

  const int half = (lane >= 16) ? 1 : 0;
#pragma unroll
  for (int tt = 0; tt < 8; ++tt) {
    const int n = colBlock + nhalf + tt * 16 + (lane & 15);
    const float b2 = 2.0f * bias[n];
#pragma unroll
    for (int i = 0; i < 8; ++i) {
      const int m = rowBlock + mbase + i + 8 * half;
      float y = acc[tt][i] + b2;
      out[(size_t)m * UNITS + n] = (y > 0.f) ? y : 0.f;
    }
  }
}

extern "C" void kernel_launch(void* const* d_in, const int* in_sizes, int n_in,
                              void* d_out, int out_size, void* d_ws, size_t ws_size,
                              hipStream_t stream) {
  const float* inputs  = (const float*)d_in[0];
  const float* context = (const float*)d_in[1];
  const float* U       = (const float*)d_in[2];
  const float* S       = (const float*)d_in[3];
  const float* V       = (const float*)d_in[4];
  const float* W       = (const float*)d_in[5];
  const float* B       = (const float*)d_in[6];
  const float* bias    = (const float*)d_in[7];
  float* out = (float*)d_out;

  char* w = (char*)d_ws;
  float*  s_ws = (float*)(w);                          // 4096*256 f32  = 4 MB
  __bf16* t_ws = (__bf16*)(w + 4u * 1024 * 1024);      // 4096*256 bf16 = 2 MB
  __bf16* Ut   = (__bf16*)(w + 6u * 1024 * 1024);      // 256 x 4096 bf16 = 2 MB
  __bf16* Vb   = (__bf16*)(w + 8u * 1024 * 1024);      // 4096 x 256 bf16 = 2 MB
  __bf16* Wt   = (__bf16*)(w + 10u * 1024 * 1024);     // 256 x 512 bf16 = 256 KB

  pack_weights<<<(N_IN * RANK) / 256, 256, 0, stream>>>(U, W, V, Ut, Wt, Vb);
  s_kernel<<<B_SZ / 64, 256, 0, stream>>>(context, Wt, B, S, s_ws);
  t_kernel<<<B_SZ / 64, 256, 0, stream>>>(inputs, Ut, s_ws, t_ws);
  out_kernel<<<dim3(UNITS / 256, B_SZ / 64), 256, 0, stream>>>(t_ws, Vb, bias, out);
}
